// AttentionLayer_661424964272
// MI455X (gfx1250) — compile-verified, matrix operand order checked by reading, not run on hardware
//
#include <hip/hip_runtime.h>
#include <hip/hip_bf16.h>

typedef __attribute__((ext_vector_type(16))) __bf16 v16bf;
typedef __attribute__((ext_vector_type(8)))  __bf16 v8bf;
typedef __attribute__((ext_vector_type(8)))  float  v8f;

#define NB  4096
#define NS  200
#define ND  64

// Dynamic-LDS layout for the main kernel (~76 KB; WGP has 320 KB).
struct __align__(16) SMem {
  __bf16 w1[80 * 256];      // W1^T, N-major, K contiguous      (40960 B)
  __bf16 w2[48 * 96];       // W2^T, zero-padded N->48, K->96   ( 9216 B)
  __bf16 h1[8][16 * 96];    // per-wave layer-1 activations     (24576 B)
  float  b1s[80];
  float  b2s[48];
  float  w3s[48];
  float  score[208];
  float  red[32];
  float  b3s;
};

// ---------------------------------------------------------------------------
// Prep: W1 (256x80 f32) -> bf16 [80][256];  W2 (80x40 f32) -> bf16 [48][96]
// packed contiguously in workspace so the main kernel copies one blob to LDS.
// ---------------------------------------------------------------------------
__global__ void din_prep_kernel(const float* __restrict__ W1,
                                const float* __restrict__ W2,
                                __bf16* __restrict__ wpack) {
  const int i = blockIdx.x * blockDim.x + threadIdx.x;
  const int n1 = 80 * 256;
  const int n2 = 48 * 96;
  if (i < n1) {
    const int n = i >> 8;
    const int k = i & 255;
    wpack[i] = (__bf16)W1[k * 80 + n];
  } else if (i < n1 + n2) {
    const int j = i - n1;
    const int n = j / 96;
    const int k = j % 96;
    const float v = (n < 40 && k < 80) ? W2[k * 40 + n] : 0.0f;
    wpack[i] = (__bf16)v;
  }
}

__device__ inline v16bf cvt16(const float* x) {
  v16bf r;
  #pragma unroll
  for (int j = 0; j < 16; ++j) r[j] = (__bf16)x[j];
  return r;
}

__device__ inline v16bf ldb(const __bf16* p) {   // LDS-resident B/A fragment
  v8bf lo = *(const v8bf*)p;
  v8bf hi = *(const v8bf*)(p + 16);
  return __builtin_shufflevector(lo, hi, 0,1,2,3,4,5,6,7,8,9,10,11,12,13,14,15);
}

// ---------------------------------------------------------------------------
// Main kernel: one block (8 wave32) per batch row.
// ---------------------------------------------------------------------------
__launch_bounds__(256, 2)
__global__ void din_attn_kernel(const float* __restrict__ seq,
                                const float* __restrict__ tgt,
                                const int*   __restrict__ mask,
                                const __bf16* __restrict__ wpack,
                                const float* __restrict__ b1,
                                const float* __restrict__ b2,
                                const float* __restrict__ W3,
                                const float* __restrict__ b3,
                                float* __restrict__ out) {
  extern __shared__ char smem_raw[];
  SMem& sm = *reinterpret_cast<SMem*>(smem_raw);

  const int b    = blockIdx.x;
  const int tid  = threadIdx.x;
  const int lane = tid & 31;
  const int w    = tid >> 5;

  // ---- cooperative LDS init: weights blob + biases ----
  {
    const uint4* src = (const uint4*)wpack;          // 50176 B = 3136 uint4
    uint4*       dst = (uint4*)&sm.w1[0];
    for (int i = tid; i < 3136; i += 256) dst[i] = src[i];
  }
  if (tid < 80) sm.b1s[tid] = b1[tid];
  if (tid < 48) sm.b2s[tid] = (tid < 40) ? b2[tid] : 0.0f;
  if (tid >= 64 && tid < 112) { const int n = tid - 64; sm.w3s[n] = (n < 40) ? W3[n] : 0.0f; }
  if (tid == 128) sm.b3s = b3[0];
  __syncthreads();

  const int mrow  = lane & 15;   // A-layout lane -> M row / B-layout lane -> N col
  const int hi16  = lane >> 4;
  const int kbase = hi16 << 3;   // K offset 0 or 8 per WMMA VGPR layout

  // ---- persistent target chunks: this lane only ever touches these 32 ----
  float tv[32];
  {
    const float* tg = tgt + (size_t)b * ND;
    #pragma unroll
    for (int q = 0; q < 4; ++q) {
      float4 a = *(const float4*)&tg[q * 16 + kbase];
      float4 c = *(const float4*)&tg[q * 16 + kbase + 4];
      tv[q*8+0]=a.x; tv[q*8+1]=a.y; tv[q*8+2]=a.z; tv[q*8+3]=a.w;
      tv[q*8+4]=c.x; tv[q*8+5]=c.y; tv[q*8+6]=c.z; tv[q*8+7]=c.w;
    }
  }

  const v8f vzero = {0.0f,0.0f,0.0f,0.0f,0.0f,0.0f,0.0f,0.0f};

  for (int mt = w; mt < 13; mt += 8) {
    const int  sidx = mt * 16 + mrow;
    const bool sok  = (sidx < NS);
    const float* srow = seq + ((size_t)b * NS + (sok ? sidx : 0)) * ND;

    // per-tile sequence chunks (the only HBM traffic of the inner loop)
    float sv[32];
    #pragma unroll
    for (int q = 0; q < 4; ++q) {
      float4 a = {0,0,0,0}, c = {0,0,0,0};
      if (sok) {
        a = *(const float4*)&srow[q * 16 + kbase];
        c = *(const float4*)&srow[q * 16 + kbase + 4];
      }
      sv[q*8+0]=a.x; sv[q*8+1]=a.y; sv[q*8+2]=a.z; sv[q*8+3]=a.w;
      sv[q*8+4]=c.x; sv[q*8+5]=c.y; sv[q*8+6]=c.z; sv[q*8+7]=c.w;
    }

    // =================== layer 1: [16x256] x [256x80] =====================
    v8f acc[5];
    #pragma unroll
    for (int i = 0; i < 5; ++i) acc[i] = vzero;

    #pragma unroll
    for (int kk = 0; kk < 8; ++kk) {             // din sections map to k-steps
      float x[16];
      #pragma unroll
      for (int j = 0; j < 16; ++j) {
        const int idx = (kk & 1) * 16 + j;
        x[j] = (kk < 2) ? tv[idx]
             : (kk < 4) ? sv[idx]
             : (kk < 6) ? (tv[idx] - sv[idx])
                        : (tv[idx] * sv[idx]);
      }
      const v16bf af = cvt16(x);
      const int   k0 = kk * 32;
      const __bf16* bbase = &sm.w1[mrow * 256 + k0 + kbase];
      #pragma unroll
      for (int nt = 0; nt < 5; ++nt) {
        const v16bf bf = ldb(bbase + nt * 16 * 256);
        acc[nt] = __builtin_amdgcn_wmma_f32_16x16x32_bf16(
            false, af, false, bf, (short)0, acc[nt], false, false);
      }
    }

    // bias + relu, re-stripe C-layout -> A-layout via per-wave LDS tile
    __bf16* h1 = &sm.h1[w][0];
    #pragma unroll
    for (int nt = 0; nt < 5; ++nt) {
      const int   n  = nt * 16 + mrow;
      const float bn = sm.b1s[n];
      #pragma unroll
      for (int r = 0; r < 8; ++r) {
        const int m = r + kbase;
        float h = fmaxf(acc[nt][r] + bn, 0.0f);
        h1[m * 96 + n] = (__bf16)h;
      }
    }
    #pragma unroll
    for (int r = 0; r < 8; ++r) {                // zero K pad 80..95
      h1[(r * 2 + hi16) * 96 + 80 + mrow] = (__bf16)0.0f;
    }

    // =================== layer 2: [16x96] x [96x48] =======================
    v8f acc2[3];
    #pragma unroll
    for (int i = 0; i < 3; ++i) acc2[i] = vzero;

    #pragma unroll
    for (int kk = 0; kk < 3; ++kk) {
      const int k0 = kk * 32;
      const v16bf af = ldb(h1 + mrow * 96 + k0 + kbase);
      #pragma unroll
      for (int nt = 0; nt < 3; ++nt) {
        const v16bf bf = ldb(&sm.w2[(nt * 16 + mrow) * 96 + k0 + kbase]);
        acc2[nt] = __builtin_amdgcn_wmma_f32_16x16x32_bf16(
            false, af, false, bf, (short)0, acc2[nt], false, false);
      }
    }

    // ============ layer 3: relu + dot(W3) + lane reduction ================
    float part[8];
    #pragma unroll
    for (int r = 0; r < 8; ++r) part[r] = 0.0f;
    #pragma unroll
    for (int nt = 0; nt < 3; ++nt) {
      const int   n   = nt * 16 + mrow;
      const float w3n = sm.w3s[n];
      const float b2n = sm.b2s[n];
      #pragma unroll
      for (int r = 0; r < 8; ++r) {
        const float h = fmaxf(acc2[nt][r] + b2n, 0.0f);
        part[r] += h * w3n;
      }
    }
    #pragma unroll
    for (int r = 0; r < 8; ++r) {
      float v = part[r];
      v += __shfl_xor(v, 1, 32);
      v += __shfl_xor(v, 2, 32);
      v += __shfl_xor(v, 4, 32);
      v += __shfl_xor(v, 8, 32);
      const int m = r + kbase;
      const int s = mt * 16 + m;
      if (mrow == 0 && s < NS) {
        const int mk = mask[b * NS + s];
        sm.score[s] = mk ? (v + sm.b3s) : -1.0e9f;
      }
    }
  }

  // ====================== masked softmax over S=200 ========================
  __syncthreads();
  const float val = (tid < NS) ? sm.score[tid] : -3.0e38f;
  float mx = val;
  mx = fmaxf(mx, __shfl_xor(mx, 16, 32));
  mx = fmaxf(mx, __shfl_xor(mx, 8, 32));
  mx = fmaxf(mx, __shfl_xor(mx, 4, 32));
  mx = fmaxf(mx, __shfl_xor(mx, 2, 32));
  mx = fmaxf(mx, __shfl_xor(mx, 1, 32));
  if (lane == 0) sm.red[w] = mx;
  __syncthreads();
  if (tid == 0) {
    float g = sm.red[0];
    #pragma unroll
    for (int i = 1; i < 8; ++i) g = fmaxf(g, sm.red[i]);
    sm.red[16] = g;
  }
  __syncthreads();
  const float gmax = sm.red[16];
  const float e = (tid < NS) ? __expf(val - gmax) : 0.0f;
  float sum = e;
  sum += __shfl_xor(sum, 16, 32);
  sum += __shfl_xor(sum, 8, 32);
  sum += __shfl_xor(sum, 4, 32);
  sum += __shfl_xor(sum, 2, 32);
  sum += __shfl_xor(sum, 1, 32);
  if (lane == 0) sm.red[8 + w] = sum;
  __syncthreads();
  if (tid == 0) {
    float g = 0.0f;
    #pragma unroll
    for (int i = 0; i < 8; ++i) g += sm.red[8 + i];
    sm.red[17] = g;
  }
  __syncthreads();
  const float inv = 1.0f / sm.red[17];
  if (tid < NS) out[(size_t)b * NS + tid] = e * inv;
}

// ---------------------------------------------------------------------------
extern "C" void kernel_launch(void* const* d_in, const int* in_sizes, int n_in,
                              void* d_out, int out_size, void* d_ws, size_t ws_size,
                              hipStream_t stream) {
  (void)in_sizes; (void)n_in; (void)out_size; (void)ws_size;
  const float* seq  = (const float*)d_in[0];
  const float* tgt  = (const float*)d_in[1];
  const int*   mask = (const int*)d_in[2];
  const float* W1   = (const float*)d_in[3];
  const float* b1   = (const float*)d_in[4];
  const float* W2   = (const float*)d_in[5];
  const float* b2   = (const float*)d_in[6];
  const float* W3   = (const float*)d_in[7];
  const float* b3   = (const float*)d_in[8];
  float* out = (float*)d_out;

  __bf16* wpack = (__bf16*)d_ws;               // 80*256 + 48*96 bf16, contiguous

  const int prep_n = 80 * 256 + 48 * 96;
  din_prep_kernel<<<(prep_n + 255) / 256, 256, 0, stream>>>(W1, W2, wpack);

  const int shmem = (int)sizeof(SMem);
  hipFuncSetAttribute((const void*)din_attn_kernel,
                      hipFuncAttributeMaxDynamicSharedMemorySize, shmem);
  din_attn_kernel<<<NB, 256, shmem, stream>>>(seq, tgt, mask, wpack,
                                              b1, b2, W3, b3, out);
}
